// Decoder_7722351198980
// MI455X (gfx1250) — matrix operand order repaired and checked by reading, not compiled, not run on hardware
//
#include <hip/hip_runtime.h>
#include <cmath>

// ---------------------------------------------------------------------------
// MI455X (gfx1250) decoder block.
// HBM-bound on the 1.6 GB fp32 rel tensors (~70us floor @ 23.3 TB/s); all
// dense math runs through v_wmma_f32_16x16x32_bf16 with f32 accumulation.
// Workgroup tile 128x64 (8 waves, 32x32 per wave -> 4 WMMAs per K-step);
// global_prefetch of the next K-tile overlaps HBM with the WMMA burst.
// ---------------------------------------------------------------------------

typedef __bf16 bf16_t;
typedef bf16_t v16bf __attribute__((ext_vector_type(16)));
typedef float  v8f   __attribute__((ext_vector_type(8)));
typedef float  v4f   __attribute__((ext_vector_type(4)));
typedef unsigned short v8us  __attribute__((ext_vector_type(8)));
typedef unsigned short v16us __attribute__((ext_vector_type(16)));

__device__ __forceinline__ unsigned short f2bf(float f) {
  return __builtin_bit_cast(unsigned short, (bf16_t)f);   // native bf16 convert
}

#define LSTR 40   // LDS row stride (ushorts): 16B-aligned chunks, conflict-free

__device__ __forceinline__ v8us cvt8(v4f f0, v4f f1) {
  v8us h;
  h[0]=f2bf(f0[0]); h[1]=f2bf(f0[1]); h[2]=f2bf(f0[2]); h[3]=f2bf(f0[3]);
  h[4]=f2bf(f1[0]); h[5]=f2bf(f1[1]); h[6]=f2bf(f1[2]); h[7]=f2bf(f1[3]);
  return h;
}

// A fragment, ISA 16-bit A 16x32 layout (lane row = lane&15, halves interleave K)
__device__ __forceinline__ v16bf load_afrag(const unsigned short* sA, int tile,
                                            int lrow, int half) {
  const unsigned short* pa = &sA[(tile * 16 + lrow) * LSTR];
  union { v16us u; v8us h[2]; } ua;
  ua.h[0] = *(const v8us*)(pa + half * 8);        // e0..7  : K = 8*half + e
  ua.h[1] = *(const v8us*)(pa + 16 + half * 8);   // e8..15 : K = 16 + 8*half + (e-8)
  return __builtin_bit_cast(v16bf, ua.u);
}

// B fragment: e0..7 K = 8*half + e, e8..15 K = 16 + 8*half + (e-8), col = lane&15
__device__ __forceinline__ v16bf load_bfrag(const unsigned short* sB, int tile,
                                            int lrow, int half) {
  const unsigned short* pb = &sB[(tile * 16 + lrow) * LSTR];
  union { v16us u; v8us h[2]; } ub;
  ub.h[0] = *(const v8us*)(pb + half * 8);
  ub.h[1] = *(const v8us*)(pb + 16 + half * 8);
  return __builtin_bit_cast(v16bf, ub.u);
}

// Generic batched GEMM:  C[z] = scale * A[z][M,K] @ B[z]  (+C_old)(+bias)(+resid)
//   b_is_nk==1 : B stored row-major [N,K]  (weight-style, x @ W.T ; K-matrix of Q@K^T)
//   b_is_nk==0 : B stored row-major [K,N]  (P @ V)
__global__ __launch_bounds__(256)
void gemm_bf16_wmma(const float* __restrict__ A, long lda, long sAb,
                    const float* __restrict__ Bm, long ldb, long sBb, int b_is_nk,
                    float* __restrict__ C, long ldc, long sCb,
                    const float* __restrict__ bias,
                    const float* __restrict__ resid, long ldr,
                    int M, int N, int K, float scale, int accumC)
{
  __shared__ unsigned short sA[128 * LSTR];
  __shared__ unsigned short sB[64 * LSTR];

  const int tid = threadIdx.x;
  const int m0 = blockIdx.y * 128;
  const int n0 = blockIdx.x * 64;
  A  += (long)blockIdx.z * sAb;
  Bm += (long)blockIdx.z * sBb;
  C  += (long)blockIdx.z * sCb;

  const int wave = tid >> 5;
  const int lane = tid & 31;
  const int mt2  = (wave & 3) * 2;    // wave's first M tile (rows (wave&3)*32 ..)
  const int ntg  = (wave >> 2) * 2;   // wave's first N tile
  const int lrow = lane & 15;
  const int half = lane >> 4;

  v8f acc00 = {}, acc01 = {}, acc10 = {}, acc11 = {};

  // staging maps
  const int srow = tid >> 2;          // 0..63
  const int scol = (tid & 3) * 8;     // 0,8,16,24
  const int tk   = tid >> 3;          // 0..31   (transposed-B staging)
  const int tn   = (tid & 7) * 8;     // 0..56

  for (int k0 = 0; k0 < K; k0 += 32) {
    // ---- stage A tile [128 x 32] f32 -> bf16 ----
#pragma unroll
    for (int rr = 0; rr < 2; ++rr) {
      v4f f0 = {}, f1 = {};
      const int r = rr * 64 + srow;
      const int gr = m0 + r;
      if (gr < M) {
        const float* src = A + (long)gr * lda + k0 + scol;
        __builtin_prefetch(src + 32, 0, 1);          // next K-tile -> global_prefetch
        f0 = *(const v4f*)(src);
        f1 = *(const v4f*)(src + 4);
      }
      *(v8us*)&sA[r * LSTR + scol] = cvt8(f0, f1);
    }
    // ---- stage B tile into sB[n][k] ----
    if (b_is_nk) {
      const float* src = Bm + (long)(n0 + srow) * ldb + k0 + scol;
      __builtin_prefetch(src + 32, 0, 1);
      v4f f0 = *(const v4f*)(src);
      v4f f1 = *(const v4f*)(src + 4);
      *(v8us*)&sB[srow * LSTR + scol] = cvt8(f0, f1);
    } else {
      // B stored [K,N]: read coalesced along n, transpose into LDS
      const float* src = Bm + (long)(k0 + tk) * ldb + n0 + tn;
      __builtin_prefetch(src + 32 * ldb, 0, 1);
      v4f f0 = *(const v4f*)(src);
      v4f f1 = *(const v4f*)(src + 4);
#pragma unroll
      for (int j = 0; j < 4; ++j) sB[(tn + j)     * LSTR + tk] = f2bf(f0[j]);
#pragma unroll
      for (int j = 0; j < 4; ++j) sB[(tn + 4 + j) * LSTR + tk] = f2bf(f1[j]);
    }
    __syncthreads();

    const v16bf a0 = load_afrag(sA, mt2,     lrow, half);
    const v16bf a1 = load_afrag(sA, mt2 + 1, lrow, half);
    const v16bf b0 = load_bfrag(sB, ntg,     lrow, half);
    const v16bf b1 = load_bfrag(sB, ntg + 1, lrow, half);

    acc00 = __builtin_amdgcn_wmma_f32_16x16x32_bf16(false, a0, false, b0,
                                                    (short)0, acc00, false, false);
    acc01 = __builtin_amdgcn_wmma_f32_16x16x32_bf16(false, a0, false, b1,
                                                    (short)0, acc01, false, false);
    acc10 = __builtin_amdgcn_wmma_f32_16x16x32_bf16(false, a1, false, b0,
                                                    (short)0, acc10, false, false);
    acc11 = __builtin_amdgcn_wmma_f32_16x16x32_bf16(false, a1, false, b1,
                                                    (short)0, acc11, false, false);
    __syncthreads();
  }

  // ---- epilogue: scale, optional C-accumulate, bias, residual ----
#pragma unroll
  for (int ti = 0; ti < 2; ++ti) {
#pragma unroll
    for (int tj = 0; tj < 2; ++tj) {
      const v8f& acc = ti ? (tj ? acc11 : acc10) : (tj ? acc01 : acc00);
      const int gc = n0 + (ntg + tj) * 16 + lrow;
      const float bv = bias ? bias[gc] : 0.0f;
#pragma unroll
      for (int i = 0; i < 8; ++i) {
        const int gr = m0 + (mt2 + ti) * 16 + i + 8 * half;
        if (gr < M) {
          const long off = (long)gr * ldc + gc;
          float y = acc[i] * scale + bv;
          if (accumC) y += C[off];
          if (resid)  y += resid[(long)gr * ldr + gc];
          C[off] = y;
        }
      }
    }
  }
}

// ---------------------------------------------------------------------------
__global__ __launch_bounds__(256)
void rmsnorm_k(const float* __restrict__ x, const float* __restrict__ w,
               float* __restrict__ o, int D)
{
  __shared__ float red[256];
  const int tid = threadIdx.x;
  const long base = (long)blockIdx.x * D;
  float ss = 0.0f;
  for (int c = tid; c < D; c += 256) { float v = x[base + c]; ss += v * v; }
  red[tid] = ss; __syncthreads();
  for (int s = 128; s > 0; s >>= 1) {
    if (tid < s) red[tid] += red[tid + s];
    __syncthreads();
  }
  const float scale = rsqrtf(red[0] / (float)D + 1e-5f);
  for (int c = tid; c < D; c += 256) o[base + c] = x[base + c] * scale * w[c];
}

// one block per (b,t) row; T == 512 (2 elements per thread)
__global__ __launch_bounds__(256)
void softmax_mask_k(float* __restrict__ sc, int T, int masked)
{
  __shared__ float red[256];
  const int tid = threadIdx.x;
  const long row = blockIdx.x;
  const int t = (int)(row % T);
  float* p = sc + row * (long)T;
  const int limit = masked ? t : (T - 1);

  float v0 = (tid       <= limit) ? p[tid]       : -__builtin_inff();
  float v1 = (tid + 256 <= limit) ? p[tid + 256] : -__builtin_inff();

  red[tid] = fmaxf(v0, v1); __syncthreads();
  for (int s = 128; s > 0; s >>= 1) {
    if (tid < s) red[tid] = fmaxf(red[tid], red[tid + s]);
    __syncthreads();
  }
  const float m = red[0]; __syncthreads();

  const float e0 = expf(v0 - m);
  const float e1 = expf(v1 - m);
  red[tid] = e0 + e1; __syncthreads();
  for (int s = 128; s > 0; s >>= 1) {
    if (tid < s) red[tid] += red[tid + s];
    __syncthreads();
  }
  const float inv = 1.0f / red[0];
  p[tid]       = e0 * inv;
  p[tid + 256] = e1 * inv;
}

__global__ __launch_bounds__(256)
void silu_mul_k(const float* __restrict__ g1, const float* __restrict__ g2,
                float* __restrict__ s, long n)
{
  const long i = (long)blockIdx.x * 256 + threadIdx.x;
  if (i < n) {
    const float a = g1[i];
    s[i] = (a / (1.0f + expf(-a))) * g2[i];
  }
}

// ---------------------------------------------------------------------------
static inline void launch_gemm(hipStream_t stream,
    const float* A, long lda, long sAb,
    const float* Bm, long ldb, long sBb, int b_is_nk,
    float* C, long ldc, long sCb,
    const float* bias, const float* resid, long ldr,
    int M, int N, int K, float scale, int accumC, int batch)
{
  dim3 g(N / 64, (M + 127) / 128, batch);
  gemm_bf16_wmma<<<g, 256, 0, stream>>>(A, lda, sAb, Bm, ldb, sBb, b_is_nk,
                                        C, ldc, sCb, bias, resid, ldr,
                                        M, N, K, scale, accumC);
}

struct AttnW {
  const float *wq, *bq, *wk, *bk, *wv, *bv, *rel, *wp, *bp;
};

extern "C" void kernel_launch(void* const* d_in, const int* in_sizes, int n_in,
                              void* d_out, int out_size, void* d_ws, size_t ws_size,
                              hipStream_t stream)
{
  (void)in_sizes; (void)n_in; (void)out_size; (void)ws_size;

  const int  B = 8, T = 512, D = 768;
  const int  MBT = B * T;                 // 4096 flattened rows
  const long BTD = (long)B * T * D;       // 3,145,728
  const long TT  = (long)T * T;

  const float* x  = (const float*)d_in[0];
  const float* nw = (const float*)d_in[1];
  AttnW mA = { (const float*)d_in[2], (const float*)d_in[3], (const float*)d_in[4],
               (const float*)d_in[5], (const float*)d_in[6], (const float*)d_in[7],
               (const float*)d_in[8], (const float*)d_in[9], (const float*)d_in[10] };
  AttnW cA = { (const float*)d_in[11], (const float*)d_in[12], (const float*)d_in[13],
               (const float*)d_in[14], (const float*)d_in[15], (const float*)d_in[16],
               (const float*)d_in[17], (const float*)d_in[18], (const float*)d_in[19] };
  const float* f_w1 = (const float*)d_in[20];
  const float* f_w2 = (const float*)d_in[21];

  float* ws    = (float*)d_ws;
  float* x_cur = ws;                // [B,T,D] running residual state
  float* xn    = ws + 1 * BTD;      // rmsnorm output
  float* qb    = ws + 2 * BTD;
  float* kb    = ws + 3 * BTD;
  float* vb    = ws + 4 * BTD;
  float* ao    = ws + 5 * BTD;      // attention context / swiglu output
  float* sc    = ws + 6 * BTD;      // scores [B,T,T]

  const float SQ = sqrtf((float)D); // source multiplies by sqrt(d), replicated

  auto attn = [&](const float* xin, float* xout, const AttnW& W, int masked) {
    rmsnorm_k<<<MBT, 256, 0, stream>>>(xin, nw, xn, D);
    // Q,K,V projections (flattened [B*T, D] @ W.T + b)
    launch_gemm(stream, xn, D, 0, W.wq, D, 0, 1, qb, D, 0, W.bq, nullptr, 0,
                MBT, D, D, 1.0f, 0, 1);
    launch_gemm(stream, xn, D, 0, W.wk, D, 0, 1, kb, D, 0, W.bk, nullptr, 0,
                MBT, D, D, 1.0f, 0, 1);
    launch_gemm(stream, xn, D, 0, W.wv, D, 0, 1, vb, D, 0, W.bv, nullptr, 0,
                MBT, D, D, 1.0f, 0, 1);
    // scores = sqrt(D) * Q @ K^T   (batched over b; K is [T,D] => weight-style)
    launch_gemm(stream, qb, D, (long)T * D, kb, D, (long)T * D, 1,
                sc, T, TT, nullptr, nullptr, 0, T, T, D, SQ, 0, B);
    // scores[:,t,:] += Q[:,t,:] @ rel[t]^T   (batched over t; streams rel once)
    launch_gemm(stream, qb, (long)T * D, D, W.rel, D, (long)T * D, 1,
                sc, TT, T, nullptr, nullptr, 0, B, T, D, 1.0f, 1, T);
    // mask (+softmax); mask-after-rel-add == reference (-inf + finite = -inf)
    softmax_mask_k<<<B * T, 256, 0, stream>>>(sc, T, masked);
    // context = P @ V   (V stored [K=T, N=D] => transposed staging path)
    launch_gemm(stream, sc, T, TT, vb, D, (long)T * D, 0,
                ao, D, (long)T * D, nullptr, nullptr, 0, T, D, T, 1.0f, 0, B);
    // out = context @ Wp^T + bp + residual(xin)
    launch_gemm(stream, ao, D, 0, W.wp, D, 0, 1, xout, D, 0, W.bp, xin, D,
                MBT, D, D, 1.0f, 0, 1);
  };

  attn(x,     x_cur, mA, 1);   // masked attention + residual
  attn(x_cur, x_cur, cA, 0);   // unmasked attention + residual (in-place safe)

  // FFN: h = xn@w1.T ; s = silu(h@w1.T) * (h@w2.T) ; out = s@w2.T + residual
  rmsnorm_k<<<MBT, 256, 0, stream>>>(x_cur, nw, xn, D);
  launch_gemm(stream, xn, D, 0, f_w1, D, 0, 1, qb, D, 0, nullptr, nullptr, 0,
              MBT, D, D, 1.0f, 0, 1);                       // h
  launch_gemm(stream, qb, D, 0, f_w1, D, 0, 1, kb, D, 0, nullptr, nullptr, 0,
              MBT, D, D, 1.0f, 0, 1);                       // g1 = h@w1.T
  launch_gemm(stream, qb, D, 0, f_w2, D, 0, 1, vb, D, 0, nullptr, nullptr, 0,
              MBT, D, D, 1.0f, 0, 1);                       // g2 = h@w2.T
  silu_mul_k<<<(unsigned)((BTD + 255) / 256), 256, 0, stream>>>(kb, vb, ao, BTD);
  launch_gemm(stream, ao, D, 0, f_w2, D, 0, 1, (float*)d_out, D, 0,
              nullptr, x_cur, D, MBT, D, D, 1.0f, 0, 1);    // out + residual
}
